// TransformerBlock_4DFTB_19782619365719
// MI455X (gfx1250) — matrix operand order repaired and checked by reading, not compiled, not run on hardware
//
#include <hip/hip_runtime.h>
#include <cstddef>
#include <cstdint>

// ---------------------------------------------------------------------------
// CDNA5 / gfx1250 types
// ---------------------------------------------------------------------------
typedef __attribute__((ext_vector_type(16))) __bf16 v16bf;
typedef __attribute__((ext_vector_type(8)))  float  v8f;
typedef __attribute__((ext_vector_type(4)))  unsigned u32x4;
typedef __attribute__((ext_vector_type(2)))  unsigned u32x2;
typedef __attribute__((ext_vector_type(4)))  float  f32x4;

union Frag16 { u32x4 q[2]; v16bf v; };   // 32B: one WMMA 16-bit operand
union Pack4  { u32x2 d;    __bf16 e[4]; };

// Problem constants (from reference)
constexpr int Bc = 2, Cc = 128, Hc = 96, Wc_ = 96, HWc = Hc * Wc_;
constexpr int HEADSc = 8, CHc = Cc / HEADSc, TOKc = 64, DKEYc = 512, HIDc = 340;
constexpr int BMt = 128, BNt = 128, BKt = 32;   // GEMM block tile
constexpr int LDA = 40, LDB = 40;               // LDS row strides (bf16 units)

__device__ __forceinline__ float seluf(float x) {
  const float al = 1.6732632423543772f, sc = 1.0507009873554805f;
  return x > 0.f ? sc * x : sc * al * (expf(x) - 1.f);
}
__device__ __forceinline__ float geluf(float x) {
  return 0.5f * x * (1.f + erff(x * 0.7071067811865475f));
}
template <int ACT> __device__ __forceinline__ float actf(float x) {
  if constexpr (ACT == 1) return seluf(x);
  else if constexpr (ACT == 2) return geluf(x);
  else return x;
}

// ---------------------------------------------------------------------------
// WMMA GEMM:  Y[b][m][p] = act( sum_k A[m][k] * X[b][k][p] + bias[m] ) (+Res)
//   IM2COL=0 : X addressed as X[k*HW + p]           (1x1 conv)
//   IM2COL=1 : k = ci*9 + 3x3 tap, zero padding     (3x3 conv, implicit GEMM)
// Block: 256 thr = 8 waves (4 along M x 2 along N); wave tile 32x64 =
// 2x4 fragments of v_wmma_f32_16x16x32_bf16 (fp32 accumulate).
// LDS: A row-major [m][k]; B transposed [n][k] so every lane fragment is
// two contiguous 16B runs -> ds_load_b128, no sub-register repacking.
// ---------------------------------------------------------------------------
template <int ACT, int IM2COL>
__global__ __launch_bounds__(256)
void gemm_wmma(const float* __restrict__ Aw, const float* __restrict__ Xin,
               const float* __restrict__ bias, float* __restrict__ Y,
               const float* __restrict__ Res,
               int M, int K, long sXb, long sYb, long sRb) {
  constexpr int P = HWc;
  __shared__ __bf16 sA [BMt][LDA];
  __shared__ __bf16 sBt[BNt][LDB];   // transposed: [n][k]

  const int tid = threadIdx.x;
  const int lane = tid & 31;
  const int wid = tid >> 5;
  const int wm = wid & 3, wn = wid >> 2;
  const int bz = blockIdx.z;
  const int m0 = blockIdx.y * BMt;
  const int n0 = blockIdx.x * BNt;
  const float* Xb = Xin + (long)bz * sXb;

  // staging index precompute
  const int arow = tid >> 1, akh = (tid & 1) * 16;        // A: one row, 16 k's
  const int bqk  = (tid >> 5) * 4, bpq = (tid & 31) * 4;  // B 1x1: 4k x 4p block
  const int bcol = tid >> 1, bkh = (tid & 1) * 16;        // B im2col: one col, 16 k's
  int bh_ = 0, bw_ = 0;
  if constexpr (IM2COL) { int p = n0 + bcol; bh_ = p / Wc_; bw_ = p - bh_ * Wc_; }

  v8f acc[2][4] = {};

  for (int k0 = 0; k0 < K; k0 += BKt) {
    // ---- stage A tile: 16 contiguous k per thread, fp32 -> bf16, 2x b128 ----
    {
      int m = m0 + arow;
      float L[16];
      if (m < M && k0 + akh + 16 <= K) {
        const f32x4* src = (const f32x4*)(Aw + (long)m * K + k0 + akh);
        #pragma unroll
        for (int i = 0; i < 4; ++i) {
          f32x4 t4 = src[i];
          L[4*i+0] = t4.x; L[4*i+1] = t4.y; L[4*i+2] = t4.z; L[4*i+3] = t4.w;
        }
      } else {
        #pragma unroll
        for (int i = 0; i < 16; ++i) {
          int k = k0 + akh + i;
          L[i] = (m < M && k < K) ? Aw[(long)m * K + k] : 0.f;
        }
      }
      Frag16 fa;
      #pragma unroll
      for (int i = 0; i < 16; ++i) fa.v[i] = (__bf16)L[i];
      *(u32x4*)&sA[arow][akh]     = fa.q[0];
      *(u32x4*)&sA[arow][akh + 8] = fa.q[1];
    }
    // ---- stage B tile into transposed LDS [n][k] ----
    if constexpr (!IM2COL) {
      // 4 k-rows x 4 p-cols per thread: coalesced float4 reads, b64 stores
      f32x4 r[4];
      #pragma unroll
      for (int i = 0; i < 4; ++i) {
        int k = k0 + bqk + i;
        if (k < K) r[i] = *(const f32x4*)(Xb + (long)k * P + n0 + bpq);
        else { f32x4 z; z.x = z.y = z.z = z.w = 0.f; r[i] = z; }
      }
      #pragma unroll
      for (int j = 0; j < 4; ++j) {
        Pack4 pk;
        pk.e[0] = (__bf16)r[0][j]; pk.e[1] = (__bf16)r[1][j];
        pk.e[2] = (__bf16)r[2][j]; pk.e[3] = (__bf16)r[3][j];
        *(u32x2*)&sBt[bpq + j][bqk] = pk.d;
      }
    } else {
      // gather 16 k-taps for one pixel column; clamp+select instead of branches
      float vals[16];
      #pragma unroll
      for (int r = 0; r < 16; ++r) {
        int k = k0 + bkh + r;
        float v = 0.f;
        if (k < K) {
          int ci = k / 9, t = k - ci * 9;
          int dy = t / 3 - 1, dx = t - (t / 3) * 3 - 1;
          int hh = bh_ + dy, ww = bw_ + dx;
          bool inb = (hh >= 0) & (hh < Hc) & (ww >= 0) & (ww < Wc_);
          int hcl = min(max(hh, 0), Hc - 1), wcl = min(max(ww, 0), Wc_ - 1);
          float lv = Xb[(long)ci * P + hcl * Wc_ + wcl];
          v = inb ? lv : 0.f;
        }
        vals[r] = v;
      }
      Frag16 fb;
      #pragma unroll
      for (int i = 0; i < 16; ++i) fb.v[i] = (__bf16)vals[i];
      *(u32x4*)&sBt[bcol][bkh]     = fb.q[0];
      *(u32x4*)&sBt[bcol][bkh + 8] = fb.q[1];
    }
    // prefetch next A chunk (global_prefetch_b8)
    if (k0 + BKt < K) {
      int m = m0 + arow;
      if (m < M) __builtin_prefetch(&Aw[(long)m * K + k0 + BKt + akh], 0, 3);
    }
    __syncthreads();

    const int l15 = lane & 15;
    const int hi = lane >> 4;
    Frag16 fA[2], fB[4];
    {
      // A frag: lanes 0-15 K 0..15, lanes 16-31 K 16..31
      int kb = hi ? 16 : 0;
      #pragma unroll
      for (int fm = 0; fm < 2; ++fm) {
        const __bf16* ar = &sA[wm * 32 + fm * 16 + l15][0];
        fA[fm].q[0] = *(const u32x4*)(ar + kb);
        fA[fm].q[1] = *(const u32x4*)(ar + kb + 8);
      }
      // B frag: lanes 0-15 K 0..15, lanes 16-31 K 16..31 (contiguous in sBt row)
      int kbB = hi ? 16 : 0;
      #pragma unroll
      for (int fn = 0; fn < 4; ++fn) {
        const __bf16* br = &sBt[wn * 64 + fn * 16 + l15][0];
        fB[fn].q[0] = *(const u32x4*)(br + kbB);
        fB[fn].q[1] = *(const u32x4*)(br + kbB + 8);
      }
    }
    #pragma unroll
    for (int fm = 0; fm < 2; ++fm)
      #pragma unroll
      for (int fn = 0; fn < 4; ++fn)
        acc[fm][fn] = __builtin_amdgcn_wmma_f32_16x16x32_bf16(
            false, fA[fm].v, false, fB[fn].v, (short)0, acc[fm][fn], false, false);

    __syncthreads();
  }

  // C/D layout: VGPR r -> M = base + hi*8 + r, N = (lane&15)
  float* Yb = Y + (long)bz * sYb;
  const int l15 = lane & 15, hi = lane >> 4;
  #pragma unroll
  for (int fm = 0; fm < 2; ++fm) {
    int mbase = m0 + wm * 32 + fm * 16 + hi * 8;
    #pragma unroll
    for (int fn = 0; fn < 4; ++fn) {
      int p = n0 + wn * 64 + fn * 16 + l15;
      if (p >= P) continue;
      #pragma unroll
      for (int r = 0; r < 8; ++r) {
        int m = mbase + r;
        if (m >= M) continue;
        float v = acc[fm][fn][r] + bias[m];
        v = actf<ACT>(v);
        if (Res) v += Res[(long)bz * sRb + (long)m * P + p];
        Yb[(long)m * P + p] = v;
      }
    }
  }
}

// ---------------------------------------------------------------------------
// Elementwise / small kernels
// ---------------------------------------------------------------------------
__global__ void ln_ch_kernel(const float* __restrict__ x, const float* __restrict__ g,
                             const float* __restrict__ be, float* __restrict__ y) {
  long idx = (long)blockIdx.x * blockDim.x + threadIdx.x;
  if (idx >= (long)Bc * HWc) return;
  int b = (int)(idx / HWc), p = (int)(idx % HWc);
  const float* xb = x + (long)b * Cc * HWc + p;
  float mu = 0.f;
  for (int c = 0; c < Cc; ++c) mu += xb[(long)c * HWc];
  mu *= (1.f / Cc);
  float var = 0.f;
  for (int c = 0; c < Cc; ++c) { float d = xb[(long)c * HWc] - mu; var += d * d; }
  var *= (1.f / Cc);
  float inv = rsqrtf(var + 1e-5f);
  float* yb = y + (long)b * Cc * HWc + p;
  for (int c = 0; c < Cc; ++c) yb[(long)c * HWc] = (xb[(long)c * HWc] - mu) * inv * g[c] + be[c];
}

__global__ void ln_last_kernel(const float* __restrict__ x, float* __restrict__ y, int D) {
  int row = blockIdx.x;
  const float* xr = x + (long)row * D;
  float* yr = y + (long)row * D;
  __shared__ float ssum[256], ssq[256];
  float a = 0.f, q = 0.f;
  for (int i = threadIdx.x; i < D; i += blockDim.x) { float v = xr[i]; a += v; q += v * v; }
  ssum[threadIdx.x] = a; ssq[threadIdx.x] = q;
  __syncthreads();
  for (int s = 128; s > 0; s >>= 1) {
    if ((int)threadIdx.x < s) { ssum[threadIdx.x] += ssum[threadIdx.x + s]; ssq[threadIdx.x] += ssq[threadIdx.x + s]; }
    __syncthreads();
  }
  float mu = ssum[0] / D;
  float var = ssq[0] / D - mu * mu;
  float inv = rsqrtf(var + 1e-5f);
  for (int i = threadIdx.x; i < D; i += blockDim.x) yr[i] = (xr[i] - mu) * inv;
}

__global__ void kv_kernel(const float* __restrict__ kn, const float* __restrict__ w,
                          const float* __restrict__ bias, float* __restrict__ out) {
  int idx = blockIdx.x * blockDim.x + threadIdx.x;      // B*TOK*2C
  if (idx >= Bc * TOKc * 2 * Cc) return;
  int o = idx % (2 * Cc), bt = idx / (2 * Cc);
  const float* kr = kn + (long)bt * DKEYc;
  const float* wr = w + (long)o * DKEYc;
  float a = bias[o];
  for (int i = 0; i < DKEYc; ++i) a += kr[i] * wr[i];
  out[idx] = a;
}

__global__ void attn_kernel(const float* __restrict__ q, const float* __restrict__ kv,
                            float* __restrict__ out) {
  long idx = (long)blockIdx.x * blockDim.x + threadIdx.x;
  if (idx >= (long)Bc * HEADSc * HWc) return;
  int p = (int)(idx % HWc);
  int bh = (int)(idx / HWc);
  int hd = bh % HEADSc, b = bh / HEADSc;
  const float* qb = q + ((long)b * Cc + hd * CHc) * HWc + p;
  float qv[CHc];
  #pragma unroll
  for (int c = 0; c < CHc; ++c) qv[c] = qb[(long)c * HWc];
  const float* kvb = kv + (long)b * TOKc * 2 * Cc;
  float sc[TOKc]; float mx = -3.4e38f;
  for (int t = 0; t < TOKc; ++t) {
    const float* kp = kvb + t * 2 * Cc + hd * CHc;
    float s = 0.f;
    #pragma unroll
    for (int c = 0; c < CHc; ++c) s += qv[c] * kp[c];
    s *= 0.08838834764831845f;      // 128^-0.5
    sc[t] = s; mx = fmaxf(mx, s);
  }
  float sum = 0.f;
  for (int t = 0; t < TOKc; ++t) { sc[t] = expf(sc[t] - mx); sum += sc[t]; }
  float inv = 1.f / sum;
  float ov[CHc];
  #pragma unroll
  for (int c = 0; c < CHc; ++c) ov[c] = 0.f;
  for (int t = 0; t < TOKc; ++t) {
    const float* vp = kvb + t * 2 * Cc + Cc + hd * CHc;
    float wgt = sc[t] * inv;
    #pragma unroll
    for (int c = 0; c < CHc; ++c) ov[c] += wgt * vp[c];
  }
  float* ob = out + ((long)b * Cc + hd * CHc) * HWc + p;
  #pragma unroll
  for (int c = 0; c < CHc; ++c) ob[(long)c * HWc] = ov[c];
}

template <int ACT>
__global__ void dwconv3x3(const float* __restrict__ X, const float* __restrict__ Wt,
                          const float* __restrict__ Bs, float* __restrict__ Y,
                          int Cch, long total) {
  long idx = (long)blockIdx.x * blockDim.x + threadIdx.x;
  if (idx >= total) return;
  int p = (int)(idx % HWc);
  long bc = idx / HWc;
  int c = (int)(bc % Cch);
  int h = p / Wc_, w = p % Wc_;
  const float* xb = X + bc * (long)HWc;
  const float* wk = Wt + (long)c * 9;
  float a = Bs[c];
  #pragma unroll
  for (int dy = -1; dy <= 1; ++dy) {
    int hh = h + dy; if (hh < 0 || hh >= Hc) continue;
    #pragma unroll
    for (int dx = -1; dx <= 1; ++dx) {
      int ww = w + dx; if (ww < 0 || ww >= Wc_) continue;
      a += xb[hh * Wc_ + ww] * wk[(dy + 1) * 3 + (dx + 1)];
    }
  }
  Y[idx] = actf<ACT>(a);
}

__global__ void interleave_kernel(const float* __restrict__ img, const float* __restrict__ dmp,
                                  float* __restrict__ dmap) {
  long idx = (long)blockIdx.x * blockDim.x + threadIdx.x;
  if (idx >= (long)Bc * Cc * HWc) return;
  int p = (int)(idx % HWc);
  long bc = idx / HWc;
  int c = (int)(bc % Cc), b = (int)(bc / Cc);
  dmap[((long)b * 2 * Cc + 2 * c) * HWc + p]     = img[idx];
  dmap[((long)b * 2 * Cc + 2 * c + 1) * HWc + p] = dmp[idx];
}

__global__ void deblur_kernel(const float* __restrict__ img, const float* __restrict__ blur,
                              float* __restrict__ out) {
  long idx = (long)blockIdx.x * blockDim.x + threadIdx.x;
  if (idx >= (long)Bc * Cc * HWc) return;
  int p = (int)(idx % HWc);
  long bc = idx / HWc;
  int c = (int)(bc % Cc), b = (int)(bc / Cc);
  int h = p / Wc_, w = p % Wc_;
  const float* ib = img + bc * (long)HWc;
  const float* kb = blur + ((long)b * Cc * 25 + (long)c * 25) * HWc + p;
  float a = 0.f;
  #pragma unroll
  for (int i = 0; i < 5; ++i) {
    int hh = min(max(h + i - 2, 0), Hc - 1);
    #pragma unroll
    for (int j = 0; j < 5; ++j) {
      int ww = min(max(w + j - 2, 0), Wc_ - 1);
      a += ib[hh * Wc_ + ww] * kb[(long)(i * 5 + j) * HWc];
    }
  }
  out[idx] = a;
}

__global__ void mean_kernel(const float* __restrict__ X, float* __restrict__ out, long strideB) {
  int row = blockIdx.x; int b = row / Cc, c = row % Cc;
  const float* p = X + (long)b * strideB + (long)c * HWc;
  __shared__ float s[256];
  float a = 0.f;
  for (int i = threadIdx.x; i < HWc; i += blockDim.x) a += p[i];
  s[threadIdx.x] = a; __syncthreads();
  for (int st = 128; st > 0; st >>= 1) {
    if ((int)threadIdx.x < st) s[threadIdx.x] += s[threadIdx.x + st];
    __syncthreads();
  }
  if (threadIdx.x == 0) out[row] = s[0] * (1.f / HWc);
}

// ha0/ha1: 3x3 convs on 1x1 spatial (zero pad) == center-tap 1x1 convs
__global__ void ha_kernel(const float* __restrict__ amean, const float* __restrict__ w0,
                          const float* __restrict__ b0, const float* __restrict__ w1,
                          const float* __restrict__ b1, float* __restrict__ a2) {
  int b = blockIdx.x, t = threadIdx.x;
  __shared__ float s1[16];
  if (t < 16) {
    float a = b0[t];
    for (int i = 0; i < Cc; ++i) a += w0[(t * Cc + i) * 9 + 4] * amean[b * Cc + i];
    s1[t] = seluf(a);
  }
  __syncthreads();
  if (t < Cc) {
    float a = b1[t];
    for (int i = 0; i < 16; ++i) a += w1[(t * 16 + i) * 9 + 4] * s1[i];
    a2[b * Cc + t] = seluf(a);
  }
}

__global__ void pred_kernel(const float* __restrict__ prompt, const float* __restrict__ w0,
                            const float* __restrict__ b0, const float* __restrict__ w1,
                            const float* __restrict__ b1, float* __restrict__ pred) {
  int b = blockIdx.x, t = threadIdx.x;
  __shared__ float s1[Cc];
  __shared__ float lg[4];
  if (t < Cc) {
    float a = b0[t];
    for (int i = 0; i < Cc; ++i) a += w0[t * Cc + i] * prompt[b * Cc + i];
    s1[t] = a;
  }
  __syncthreads();
  if (t < 4) {
    float a = b1[t];
    for (int i = 0; i < Cc; ++i) a += w1[t * Cc + i] * s1[i];
    lg[t] = a;
  }
  __syncthreads();
  if (t == 0) {
    float mx = fmaxf(fmaxf(lg[0], lg[1]), fmaxf(lg[2], lg[3]));
    float e0 = expf(lg[0] - mx), e1 = expf(lg[1] - mx), e2 = expf(lg[2] - mx), e3 = expf(lg[3] - mx);
    float s = e0 + e1 + e2 + e3;
    pred[b * 4 + 0] = e0 / s; pred[b * 4 + 1] = e1 / s;
    pred[b * 4 + 2] = e2 / s; pred[b * 4 + 3] = e3 / s;
  }
}

__global__ void combine_kernel(const float* __restrict__ debl, const float* __restrict__ n2,
                               const float* __restrict__ img, const float* __restrict__ tt,
                               const float* __restrict__ m2, const float* __restrict__ ad,
                               const float* __restrict__ a2, const float* __restrict__ pred,
                               float* __restrict__ out) {
  long idx = (long)blockIdx.x * blockDim.x + threadIdx.x;
  if (idx >= (long)Bc * Cc * HWc) return;
  long bc = idx / HWc;
  int c = (int)(bc % Cc), b = (int)(bc / Cc);
  float p0 = pred[b * 4 + 0], p1 = pred[b * 4 + 1], p2 = pred[b * 4 + 2], p3 = pred[b * 4 + 3];
  float av = a2[b * Cc + c];
  float ig = img[idx];
  out[idx] = p0 * debl[idx] + p1 * (n2[idx] + ig) + p2 * (tt[idx] * (ig - av) + av)
           + p3 * (ig * m2[idx] + ad[idx]);
}

// ---------------------------------------------------------------------------
// Orchestration
// ---------------------------------------------------------------------------
extern "C" void kernel_launch(void* const* d_in, const int* in_sizes, int n_in,
                              void* d_out, int out_size, void* d_ws, size_t ws_size,
                              hipStream_t stream) {
  const float* x      = (const float*)d_in[0];
  const float* key    = (const float*)d_in[1];
  const float* prompt = (const float*)d_in[2];
  const float* prm[62];
  for (int i = 0; i < 62 && 3 + i < n_in; ++i) prm[i] = (const float*)d_in[3 + i];

  enum { i_norm1_w = 0, i_norm1_b, i_norm2_w, i_norm2_b,
         i_q0_w, i_q0_b, i_q1_w, i_q1_b, i_po_w, i_po_b,
         i_kv_w, i_kv_b, i_fc1_w, i_fc1_b, i_dw_w, i_dw_b, i_fc2_w, i_fc2_b,
         i_mlp_img_w, i_mlp_img_b, i_mlp_degra_w, i_mlp_degra_b,
         i_pred0_w, i_pred0_b, i_pred1_w, i_pred1_b,
         i_db0_w, i_db0_b, i_db1_w, i_db1_b, i_db2_w, i_db2_b, i_db3_w, i_db3_b,
         i_dn0_w, i_dn0_b, i_dn1_w, i_dn1_b, i_dn2_w, i_dn2_b,
         i_ha0_w, i_ha0_b, i_ha1_w, i_ha1_b,
         i_ht0_w, i_ht0_b, i_ht1_w, i_ht1_b, i_ht2_w, i_ht2_b,
         i_dm0_w, i_dm0_b, i_dm1_w, i_dm1_b, i_dm2_w, i_dm2_b,
         i_da0_w, i_da0_b, i_da1_w, i_da1_b, i_da2_w, i_da2_b };

  size_t off = 0;
  auto alloc = [&](long n) -> float* {
    float* r = (float*)((char*)d_ws + off);
    off += ((size_t)n * sizeof(float) + 255) & ~(size_t)255;
    return r;
  };
  const long NF = (long)Bc * Cc * HWc;
  float* xn    = alloc(NF);
  float* q0o   = alloc(NF);
  float* qb    = alloc(NF);
  float* att   = alloc(NF);
  float* degra = alloc(NF);
  float* img   = alloc(NF);
  float* dmp   = alloc(NF);
  float* dmap  = alloc(2 * NF);
  float* t     = alloc(NF);
  float* s8    = alloc((long)Bc * 16 * HWc);
  float* t2    = alloc(NF);
  float* blur  = alloc((long)Bc * Cc * 25 * HWc);
  float* debl  = alloc(NF);
  float* n2    = alloc(NF);
  float* tmpC  = alloc(NF);
  float* ttb   = alloc(NF);
  float* m2    = alloc(NF);
  float* ad    = alloc(NF);
  float* xcomb = alloc(NF);
  float* y1    = alloc(NF);
  float* y2    = alloc((long)Bc * HIDc * HWc);
  float* y3    = alloc((long)Bc * HIDc * HWc);
  float* kn    = alloc((long)Bc * TOKc * DKEYc);
  float* kvb   = alloc((long)Bc * TOKc * 2 * Cc);
  float* amean = alloc(Bc * Cc);
  float* a2    = alloc(Bc * Cc);

  float* out0  = (float*)d_out;
  float* predo = out0 + NF;   // pred output tail: B*OPS = 8 floats

  auto gemm = [&](const float* A, const float* X, const float* bias, float* Y,
                  const float* Res, int M, int K, long sX, long sY, long sR,
                  int act, int im2col) {
    dim3 g(HWc / BNt, (M + BMt - 1) / BMt, Bc), bt(256);
    if (!im2col) {
      if (act == 1) gemm_wmma<1, 0><<<g, bt, 0, stream>>>(A, X, bias, Y, Res, M, K, sX, sY, sR);
      else          gemm_wmma<0, 0><<<g, bt, 0, stream>>>(A, X, bias, Y, Res, M, K, sX, sY, sR);
    } else {
      if (act == 1) gemm_wmma<1, 1><<<g, bt, 0, stream>>>(A, X, bias, Y, Res, M, K, sX, sY, sR);
      else          gemm_wmma<0, 1><<<g, bt, 0, stream>>>(A, X, bias, Y, Res, M, K, sX, sY, sR);
    }
  };

  const long sC = (long)Cc * HWc, s2C = 2 * sC, s8s = (long)16 * HWc, sHID = (long)HIDc * HWc;
  const unsigned gNF = (unsigned)((NF + 255) / 256);
  const unsigned gPix = (unsigned)((Bc * HWc + 255) / 256);

  // ---- attention branch ----
  ln_ch_kernel<<<gPix, 256, 0, stream>>>(x, prm[i_norm1_w], prm[i_norm1_b], xn);
  gemm(prm[i_q0_w], xn, prm[i_q0_b], q0o, nullptr, Cc, Cc, sC, sC, 0, 0, 0);
  dwconv3x3<0><<<gNF, 256, 0, stream>>>(q0o, prm[i_q1_w], prm[i_q1_b], qb, Cc, NF);
  ln_last_kernel<<<Bc * TOKc, 256, 0, stream>>>(key, kn, DKEYc);
  kv_kernel<<<(Bc * TOKc * 2 * Cc + 255) / 256, 256, 0, stream>>>(kn, prm[i_kv_w], prm[i_kv_b], kvb);
  attn_kernel<<<(unsigned)(((long)Bc * HEADSc * HWc + 255) / 256), 256, 0, stream>>>(qb, kvb, att);
  gemm(prm[i_po_w], att, prm[i_po_b], degra, nullptr, Cc, Cc, sC, sC, 0, 0, 0);

  // ---- prompt -> op weights ----
  pred_kernel<<<Bc, 128, 0, stream>>>(prompt, prm[i_pred0_w], prm[i_pred0_b],
                                      prm[i_pred1_w], prm[i_pred1_b], predo);

  // ---- img / degra maps, interleaved stack ----
  gemm(prm[i_mlp_img_w], x, prm[i_mlp_img_b], img, nullptr, Cc, Cc, sC, sC, 0, 0, 0);
  gemm(prm[i_mlp_degra_w], degra, prm[i_mlp_degra_b], dmp, nullptr, Cc, Cc, sC, sC, 0, 0, 0);
  interleave_kernel<<<gNF, 256, 0, stream>>>(img, dmp, dmap);

  // ---- deblur branch (kernel prediction) ----
  gemm(prm[i_db0_w], dmap, prm[i_db0_b], t, nullptr, Cc, 2 * Cc * 9, s2C, sC, 0, 0, 1);
  gemm(prm[i_db1_w], t, prm[i_db1_b], s8, nullptr, 16, Cc * 9, sC, s8s, 0, 1, 1);
  gemm(prm[i_db2_w], s8, prm[i_db2_b], t2, nullptr, Cc, 16 * 9, s8s, sC, 0, 1, 1);
  gemm(prm[i_db3_w], t2, prm[i_db3_b], blur, nullptr, Cc * 25, Cc, sC, (long)Cc * 25 * HWc, 0, 0, 0);
  deblur_kernel<<<gNF, 256, 0, stream>>>(img, blur, debl);

  // ---- denoise branch ----
  gemm(prm[i_dn0_w], dmap + sC, prm[i_dn0_b], tmpC, nullptr, Cc, Cc * 9, s2C, sC, 0, 0, 1);
  gemm(prm[i_dn1_w], tmpC, prm[i_dn1_b], s8, nullptr, 16, Cc * 9, sC, s8s, 0, 1, 1);
  gemm(prm[i_dn2_w], s8, prm[i_dn2_b], n2, nullptr, Cc, 16 * 9, s8s, sC, 0, 1, 1);

  // ---- dehaze branch ----
  mean_kernel<<<Bc * Cc, 256, 0, stream>>>(dmap + sC, amean, s2C);
  ha_kernel<<<Bc, 128, 0, stream>>>(amean, prm[i_ha0_w], prm[i_ha0_b],
                                    prm[i_ha1_w], prm[i_ha1_b], a2);
  gemm(prm[i_ht0_w], dmap, prm[i_ht0_b], tmpC, nullptr, Cc, Cc * 9, s2C, sC, 0, 0, 1);
  gemm(prm[i_ht1_w], tmpC, prm[i_ht1_b], s8, nullptr, 16, Cc * 9, sC, s8s, 0, 1, 1);
  gemm(prm[i_ht2_w], s8, prm[i_ht2_b], ttb, nullptr, Cc, 16 * 9, s8s, sC, 0, 1, 1);

  // ---- dedark branch ----
  gemm(prm[i_dm0_w], dmap, prm[i_dm0_b], tmpC, nullptr, Cc, Cc * 9, s2C, sC, 0, 0, 1);
  gemm(prm[i_dm1_w], tmpC, prm[i_dm1_b], s8, nullptr, 16, Cc * 9, sC, s8s, 0, 1, 1);
  gemm(prm[i_dm2_w], s8, prm[i_dm2_b], m2, nullptr, Cc, 16 * 9, s8s, sC, 0, 1, 1);
  gemm(prm[i_da0_w], dmap + sC, prm[i_da0_b], tmpC, nullptr, Cc, Cc * 9, s2C, sC, 0, 0, 1);
  gemm(prm[i_da1_w], tmpC, prm[i_da1_b], s8, nullptr, 16, Cc * 9, sC, s8s, 0, 1, 1);
  gemm(prm[i_da2_w], s8, prm[i_da2_b], ad, nullptr, Cc, 16 * 9, s8s, sC, 0, 1, 1);

  // ---- weighted combine + FFN ----
  combine_kernel<<<gNF, 256, 0, stream>>>(debl, n2, img, ttb, m2, ad, a2, predo, xcomb);
  ln_ch_kernel<<<gPix, 256, 0, stream>>>(xcomb, prm[i_norm2_w], prm[i_norm2_b], y1);
  gemm(prm[i_fc1_w], y1, prm[i_fc1_b], y2, nullptr, HIDc, Cc, sC, sHID, 0, 0, 0);
  dwconv3x3<2><<<(unsigned)(((long)Bc * HIDc * HWc + 255) / 256), 256, 0, stream>>>(
      y2, prm[i_dw_w], prm[i_dw_b], y3, HIDc, (long)Bc * HIDc * HWc);
  gemm(prm[i_fc2_w], y3, prm[i_fc2_b], out0, xcomb, Cc, HIDc, sHID, sC, sC, 0, 0);
}
